// CombinedLoss_58729382806236
// MI455X (gfx1250) — matrix operand order repaired and checked by reading, not compiled, not run on hardware
//
#include <hip/hip_runtime.h>
#include <math.h>

#define BATCH 8192
#define DIM   128
#define NCLS  1024
#define NPART 4
#define EPSV  1e-6f

typedef __attribute__((ext_vector_type(16))) __bf16 v16bf;
typedef __attribute__((ext_vector_type(8)))  float  v8f;

union ABu { v16bf bf; unsigned int u[8]; uint4 q[2]; };

__device__ __forceinline__ unsigned short f2bf(float x){
    unsigned int u = __float_as_uint(x);
    u = (u + 0x7FFFu + ((u >> 16) & 1u)) >> 16;
    return (unsigned short)u;
}

__device__ __forceinline__ float wred_add(float v){
#pragma unroll
    for (int o = 16; o > 0; o >>= 1) v += __shfl_xor(v, o, 32);
    return v;
}

// ---------------- Kernel 1: normalize rows, emit fp32 + bf16 + row stats ----
__global__ void __launch_bounds__(256) k_normalize(
    const float* __restrict__ feat, float* __restrict__ fn,
    unsigned short* __restrict__ fh, float* __restrict__ rowsum,
    float* __restrict__ f2out)
{
    int wid = threadIdx.x >> 5, lane = threadIdx.x & 31;
    int i = blockIdx.x * 8 + wid;
    float4 v = ((const float4*)feat)[i * 32 + lane];
    float ss = wred_add(v.x*v.x + v.y*v.y + v.z*v.z + v.w*v.w);
    float inv = 1.0f / fmaxf(sqrtf(ss), 1e-12f);
    float4 o; o.x = v.x*inv; o.y = v.y*inv; o.z = v.z*inv; o.w = v.w*inv;
    ((float4*)fn)[i * 32 + lane] = o;
    unsigned int* fhu = (unsigned int*)fh;
    fhu[i*64 + lane*2]     = (unsigned int)f2bf(o.x) | ((unsigned int)f2bf(o.y) << 16);
    fhu[i*64 + lane*2 + 1] = (unsigned int)f2bf(o.z) | ((unsigned int)f2bf(o.w) << 16);
    float rs = wred_add(o.x + o.y + o.z + o.w);
    float sq = wred_add(o.x*o.x + o.y*o.y + o.z*o.z + o.w*o.w);
    if (lane == 0){ rowsum[i] = rs; f2out[i] = sq; }
}

// ---------------- Kernel 2: deterministic centroids (block per class) -------
__global__ void __launch_bounds__(128) k_centroid(
    const float* __restrict__ fn, const int* __restrict__ labels,
    float* __restrict__ cent, unsigned short* __restrict__ ch,
    float* __restrict__ counts, float* __restrict__ c2,
    float* __restrict__ csum)
{
    int c = blockIdx.x, t = threadIdx.x;
    float acc = 0.f, cnt = 0.f;
    for (int row = 0; row < BATCH; ++row){
        if (labels[row] == c){ acc += fn[row * DIM + t]; cnt += 1.f; }  // block-uniform branch
    }
    float val = acc / fmaxf(cnt, 1.f);
    cent[c * DIM + t] = val;
    ch[c * DIM + t]   = f2bf(val);
    __shared__ float s1[128], s2[128];
    s1[t] = val; s2[t] = val * val;
    __syncthreads();
    for (int o = 64; o > 0; o >>= 1){
        if (t < o){ s1[t] += s1[t + o]; s2[t] += s2[t + o]; }
        __syncthreads();
    }
    if (t == 0){ counts[c] = cnt; csum[c] = s1[0]; c2[c] = s2[0]; }
}

// ---- Kernel 3: WMMA gram + fused hardest-negative argmax -------------------
// Block = 128 anchor rows (8 waves x 16-row tiles). All waves share one B-tile
// staged in LDS via double-buffered GLOBAL_LOAD_ASYNC_TO_LDS_B128 (ASYNCcnt).
// gridDim.y splits the 512 column tiles into NPART partitions.
#define BSTRIDE 136   // 128 bf16 + 8 pad (row pitch 272B -> conflict-free b128)
__global__ void __launch_bounds__(256) k_hardneg(
    const unsigned short* __restrict__ fh, const int* __restrict__ labels,
    float* __restrict__ negv, int* __restrict__ negi)
{
    __shared__ int rl[128];
    __shared__ __align__(16) unsigned short Bt[2][16 * BSTRIDE];
    int tid = threadIdx.x, wid = tid >> 5, lane = tid & 31;
    int i0 = blockIdx.x * 128;
    int part = blockIdx.y;
    if (tid < 128) rl[tid] = labels[i0 + tid];

    int m = lane & 15, hi = lane >> 4;
    const unsigned int* fhu = (const unsigned int*)fh;

    // A fragments for this wave's 16 rows (contiguous k = 16*hi + s order)
    ABu a[4];
    int rowA = (i0 + wid * 16 + m) * (DIM / 2);
#pragma unroll
    for (int c = 0; c < 4; ++c){
        int kb = c * 32;
#pragma unroll
        for (int v = 0; v < 4; ++v) a[c].u[v]   = fhu[rowA + ((kb + 16*hi + 2*v) >> 1)];
#pragma unroll
        for (int v = 0; v < 4; ++v) a[c].u[v+4] = fhu[rowA + ((kb + 16*hi + 8 + 2*v) >> 1)];
    }

    // this thread's staging slot: one 16B piece of the 16x128 bf16 B tile
    int sn = tid >> 4;            // column 0..15
    int sk = (tid & 15) * 8;      // k start (8 bf16 = 16 bytes)
    unsigned ldsoff0 = (unsigned)(uintptr_t)&Bt[0][sn * BSTRIDE + sk];
    unsigned ldsoff1 = (unsigned)(uintptr_t)&Bt[1][sn * BSTRIDE + sk];
    const int jt0 = part * (BATCH / 16 / NPART);   // 128 column tiles per part

    {   // prefetch first tile into buffer 0 (async, ASYNCcnt-tracked)
        unsigned long long ga = (unsigned long long)(uintptr_t)(fh + ((jt0 * 16 + sn) * DIM + sk));
        asm volatile("global_load_async_to_lds_b128 %0, %1, off"
                     :: "v"(ldsoff0), "v"(ga) : "memory");
    }
    __syncthreads();              // rl[] visible
    int myl[8];
#pragma unroll
    for (int r = 0; r < 8; ++r) myl[r] = rl[wid * 16 + r + 8*hi];
    float bestv[8]; int besti[8];
#pragma unroll
    for (int r = 0; r < 8; ++r){ bestv[r] = -1e30f; besti[r] = 0; }

    int cur = 0;
    for (int t2 = 0; t2 < BATCH / 16 / NPART; ++t2){
        asm volatile("s_wait_asynccnt 0x0" ::: "memory");
        __syncthreads();          // current buffer fully staged by all waves
        if (t2 + 1 < BATCH / 16 / NPART){
            int jn = jt0 + t2 + 1;
            unsigned long long ga = (unsigned long long)(uintptr_t)(fh + ((jn * 16 + sn) * DIM + sk));
            unsigned dst = cur ? ldsoff0 : ldsoff1;
            asm volatile("global_load_async_to_lds_b128 %0, %1, off"
                         :: "v"(dst), "v"(ga) : "memory");
        }
        int j0 = (jt0 + t2) * 16;
        const unsigned short* bt = Bt[cur];
        v8f acc = {};
#pragma unroll
        for (int c = 0; c < 4; ++c){
            ABu b;
            const uint4* p = (const uint4*)&bt[m * BSTRIDE + c * 32 + 16*hi];
            b.q[0] = p[0];
            b.q[1] = p[1];
            acc = __builtin_amdgcn_wmma_f32_16x16x32_bf16(
                false, a[c].bf, false, b.bf, (short)0, acc, false, false);
        }
        int lj = labels[j0 + m];
        int j  = j0 + m;
#pragma unroll
        for (int r = 0; r < 8; ++r){
            float g  = acc[r];
            bool upd = (lj != myl[r]) && (g > bestv[r]);   // max gram == min d2 (diff label)
            bestv[r] = upd ? g : bestv[r];
            besti[r] = upd ? j : besti[r];
        }
        __syncthreads();          // all waves done reading buffer before restage
        cur ^= 1;
    }

    // per-row reduction across the 16 N-lanes of this wave (value + index)
#pragma unroll
    for (int r = 0; r < 8; ++r){
        float v  = bestv[r];
        int   ix = besti[r];
#pragma unroll
        for (int mk = 1; mk < 16; mk <<= 1){
            float ov = __shfl_xor(v, mk, 32);
            int   oi = __shfl_xor(ix, mk, 32);
            bool t = ov > v;
            v  = t ? ov : v;
            ix = t ? oi : ix;
        }
        if (m == 0){
            int row = i0 + wid * 16 + r + 8*hi;
            negv[part * BATCH + row] = v;
            negi[part * BATCH + row] = ix;
        }
    }
}

// ---------------- Kernel 4: WMMA f@centroids^T + fused masked min(dc2) ------
__global__ void __launch_bounds__(256) k_mincent(
    const unsigned short* __restrict__ fh, const unsigned short* __restrict__ ch,
    const int* __restrict__ labels, const float* __restrict__ counts,
    const float* __restrict__ c2, const float* __restrict__ csum,
    const float* __restrict__ f2v, const float* __restrict__ rowsum,
    float* __restrict__ min_dc2)
{
    __shared__ int   rl[16];
    __shared__ float rf2[16], rsm[16];
    __shared__ float redv[2048];
    int tid = threadIdx.x, wid = tid >> 5, lane = tid & 31;
    int i0 = blockIdx.x * 16;
    if (tid < 16){ rl[tid] = labels[i0+tid]; rf2[tid] = f2v[i0+tid]; rsm[tid] = rowsum[i0+tid]; }
    __syncthreads();
    int m = lane & 15, hi = lane >> 4;
    const unsigned int* fhu = (const unsigned int*)fh;
    const unsigned int* chu = (const unsigned int*)ch;

    ABu a[4];
    int rowA = (i0 + m) * (DIM / 2);
#pragma unroll
    for (int c = 0; c < 4; ++c){
        int kb = c * 32;
#pragma unroll
        for (int v = 0; v < 4; ++v) a[c].u[v]   = fhu[rowA + ((kb + 16*hi + 2*v) >> 1)];
#pragma unroll
        for (int v = 0; v < 4; ++v) a[c].u[v+4] = fhu[rowA + ((kb + 16*hi + 8 + 2*v) >> 1)];
    }
    int myl[8]; float myf2[8], mysm[8];
#pragma unroll
    for (int r = 0; r < 8; ++r){ myl[r] = rl[r+8*hi]; myf2[r] = rf2[r+8*hi]; mysm[r] = rsm[r+8*hi]; }
    float minv[8];
#pragma unroll
    for (int r = 0; r < 8; ++r) minv[r] = 1e30f;

    for (int jt = wid; jt < NCLS / 16; jt += 8){
        int j0 = jt * 16;
        int k  = j0 + m;                       // class index of this column
        int rowB = k * (DIM / 2);
        v8f acc = {};
#pragma unroll
        for (int c = 0; c < 4; ++c){
            ABu b;
            int kb = c * 32;
#pragma unroll
            for (int v = 0; v < 8; ++v) b.u[v] = chu[rowB + ((kb + 16*hi + 2*v) >> 1)];
            acc = __builtin_amdgcn_wmma_f32_16x16x32_bf16(
                false, a[c].bf, false, b.bf, (short)0, acc, false, false);
        }
        float ck2 = c2[k], cks = csum[k], ckn = counts[k];
        bool nonempty = (ckn != 0.0f);
#pragma unroll
        for (int r = 0; r < 8; ++r){
            float dc2 = myf2[r] + ck2 - 2.0f * acc[r]
                      + 2.0f * EPSV * (mysm[r] - cks) + (float)DIM * EPSV * EPSV;
            bool upd = nonempty && (myl[r] != k) && (dc2 < minv[r]);
            minv[r]  = upd ? dc2 : minv[r];
        }
    }
#pragma unroll
    for (int r = 0; r < 8; ++r) redv[(r + 8*hi) * 128 + wid * 16 + m] = minv[r];
    __syncthreads();
    if (tid < 16){
        float mv = 1e30f;
        for (int s = 0; s < 128; ++s) mv = fminf(mv, redv[tid * 128 + s]);
        min_dc2[i0 + tid] = mv;
    }
}

// ---------------- Kernel 5: per-anchor exact fp32 pdists + loss terms -------
__global__ void __launch_bounds__(256) k_peranchor(
    const float* __restrict__ fn, const int* __restrict__ labels,
    const int* __restrict__ pos_idx, const float* __restrict__ negv,
    const int* __restrict__ negi, const float* __restrict__ cent,
    const float* __restrict__ counts, const float* __restrict__ min_dc2,
    const int* __restrict__ epoch, float* __restrict__ ltrip,
    float* __restrict__ lcent, float* __restrict__ validf)
{
    int wid = threadIdx.x >> 5, lane = threadIdx.x & 31;
    int i = blockIdx.x * 8 + wid;
    int lab = labels[i], p = pos_idx[i];
    // merge the NPART column-partition argmax partials
    float bv = negv[i];
    int   ni = negi[i];
#pragma unroll
    for (int p2 = 1; p2 < NPART; ++p2){
        float v2 = negv[p2 * BATCH + i];
        int   i2 = negi[p2 * BATCH + i];
        bool t = v2 > bv;
        bv = t ? v2 : bv;
        ni = t ? i2 : ni;
    }
    const float4* fn4 = (const float4*)fn;
    const float4* c4  = (const float4*)cent;
    float4 fi = fn4[i * 32 + lane];
    float4 fp = fn4[p * 32 + lane];
    float4 fq = fn4[ni * 32 + lane];
    float4 co = c4[lab * 32 + lane];
    float cnt = counts[lab];
    float inv = 1.0f / fmaxf(cnt - 1.0f, 1.0f);
    float sp = 0.f, sn = 0.f, sc = 0.f, d, cp;
#define ACC(comp) \
    d = fi.comp - fp.comp + EPSV; sp += d * d; \
    d = fi.comp - fq.comp + EPSV; sn += d * d; \
    cp = (cnt * co.comp - fi.comp) * inv;      \
    d = fi.comp - cp + EPSV; sc += d * d;
    ACC(x) ACC(y) ACC(z) ACC(w)
#undef ACC
    sp = wred_add(sp); sn = wred_add(sn); sc = wred_add(sc);
    if (lane == 0){
        float d_p = sqrtf(sp), d_n = sqrtf(sn), d_pc = sqrtf(sc);
        float d_nc = sqrtf(fmaxf(min_dc2[i], 0.0f));
        float e  = (float)epoch[0];
        float tt = fminf(0.3f + 0.04f * e, 1.0f);
        float tc = fminf(0.3f + 0.02f * e, 1.0f);
        bool valid = (cnt >= 2.0f) && (cnt < (float)BATCH);
        float lt = log1pf(expf((d_p - d_n) / tt));
        float lc = log1pf(expf((d_pc - d_nc) / tc));
        ltrip[i]  = valid ? lt : 0.0f;
        lcent[i]  = valid ? lc : 0.0f;
        validf[i] = valid ? 1.0f : 0.0f;
    }
}

// ---------------- Kernel 6: deterministic single-block final reduction ------
__global__ void __launch_bounds__(256) k_final(
    const float* __restrict__ ltrip, const float* __restrict__ lcent,
    const float* __restrict__ validf, float* __restrict__ out)
{
    __shared__ float s0[256], s1[256], s2[256];
    int t = threadIdx.x;
    float a = 0.f, b = 0.f, c = 0.f;
    for (int idx = t; idx < BATCH; idx += 256){
        a += ltrip[idx]; b += lcent[idx]; c += validf[idx];
    }
    s0[t] = a; s1[t] = b; s2[t] = c;
    __syncthreads();
    for (int o = 128; o > 0; o >>= 1){
        if (t < o){ s0[t] += s0[t+o]; s1[t] += s1[t+o]; s2[t] += s2[t+o]; }
        __syncthreads();
    }
    if (t == 0){
        float nv = fmaxf(s2[0], 1.0f);
        out[0] = s0[0] / nv + 0.0005f * s1[0] / nv;
    }
}

extern "C" void kernel_launch(void* const* d_in, const int* in_sizes, int n_in,
                              void* d_out, int out_size, void* d_ws, size_t ws_size,
                              hipStream_t stream)
{
    const float* feat   = (const float*)d_in[0];
    const int*  labels  = (const int*)d_in[1];
    const int*  posidx  = (const int*)d_in[2];
    const int*  epoch   = (const int*)d_in[3];
    float* out = (float*)d_out;

    char* w = (char*)d_ws;
    auto alloc = [&](size_t bytes) -> char* {
        char* p = w; w += (bytes + 255) & ~(size_t)255; return p;
    };
    float*          fn      = (float*)         alloc((size_t)BATCH * DIM * 4);
    unsigned short* fh      = (unsigned short*)alloc((size_t)BATCH * DIM * 2);
    float*          rowsum  = (float*)         alloc((size_t)BATCH * 4);
    float*          f2v     = (float*)         alloc((size_t)BATCH * 4);
    float*          cent    = (float*)         alloc((size_t)NCLS * DIM * 4);
    unsigned short* ch      = (unsigned short*)alloc((size_t)NCLS * DIM * 2);
    float*          counts  = (float*)         alloc((size_t)NCLS * 4);
    float*          c2      = (float*)         alloc((size_t)NCLS * 4);
    float*          csum    = (float*)         alloc((size_t)NCLS * 4);
    float*          negv    = (float*)         alloc((size_t)NPART * BATCH * 4);
    int*            negi    = (int*)           alloc((size_t)NPART * BATCH * 4);
    float*          mindc2  = (float*)         alloc((size_t)BATCH * 4);
    float*          ltrip   = (float*)         alloc((size_t)BATCH * 4);
    float*          lcent   = (float*)         alloc((size_t)BATCH * 4);
    float*          validf  = (float*)         alloc((size_t)BATCH * 4);

    k_normalize<<<BATCH / 8, 256, 0, stream>>>(feat, fn, fh, rowsum, f2v);
    k_centroid <<<NCLS, 128, 0, stream>>>(fn, labels, cent, ch, counts, c2, csum);
    dim3 gHard(BATCH / 128, NPART);
    k_hardneg  <<<gHard, 256, 0, stream>>>(fh, labels, negv, negi);
    k_mincent  <<<BATCH / 16, 256, 0, stream>>>(fh, ch, labels, counts, c2, csum, f2v, rowsum, mindc2);
    k_peranchor<<<BATCH / 8, 256, 0, stream>>>(fn, labels, posidx, negv, negi, cent, counts,
                                               mindc2, epoch, ltrip, lcent, validf);
    k_final    <<<1, 256, 0, stream>>>(ltrip, lcent, validf, out);
}
